// SwitchTransformersSparseMLP_43714177139200
// MI455X (gfx1250) — matrix operand order repaired and checked by reading, not compiled, not run on hardware
//
#include <hip/hip_runtime.h>

#define B_ 8
#define S_ 2048
#define D_ 1024
#define F_ 4096
#define E_ 8
#define C_ 512
#define T_ (B_*S_)        // 16384 tokens
#define PAIRS (B_*E_)     // 64 (b,e) pairs
#define ROWS (PAIRS*C_)   // 32768 buffer rows

typedef __attribute__((ext_vector_type(16))) __bf16 v16bf;
typedef __attribute__((ext_vector_type(8)))  float  v8f;
typedef __attribute__((ext_vector_type(8)))  unsigned short v8us;
typedef __attribute__((ext_vector_type(16))) unsigned short v16us;

static __device__ __forceinline__ unsigned short f2bf(float f) {
  unsigned u = __float_as_uint(f);
  unsigned r = u + 0x7FFFu + ((u >> 16) & 1u);
  return (unsigned short)(r >> 16);
}

// ---------------- Router: logits, softmax max-prob, argmax ----------------
__global__ __launch_bounds__(256) void k_router(const float* __restrict__ hs,
                                                const float* __restrict__ rw,
                                                float* __restrict__ logits_out,
                                                int* __restrict__ expert_id,
                                                float* __restrict__ maxprob) {
  __shared__ float rwl[D_ * E_];
  for (int i = threadIdx.x; i < D_ * E_; i += 256) rwl[i] = rw[i];
  __syncthreads();
  int wave = threadIdx.x >> 5, lane = threadIdx.x & 31;
  int token = blockIdx.x * 8 + wave;
  const float* row = hs + (size_t)token * D_;
  float acc[E_];
#pragma unroll
  for (int e = 0; e < E_; ++e) acc[e] = 0.f;
  for (int j = 0; j < D_ / 32; ++j) {
    int d = j * 32 + lane;
    float h = row[d];
    const float* w = &rwl[d * E_];
#pragma unroll
    for (int e = 0; e < E_; ++e) acc[e] += h * w[e];
  }
#pragma unroll
  for (int e = 0; e < E_; ++e) {
    acc[e] += __shfl_xor(acc[e], 16, 32);
    acc[e] += __shfl_xor(acc[e], 8, 32);
    acc[e] += __shfl_xor(acc[e], 4, 32);
    acc[e] += __shfl_xor(acc[e], 2, 32);
    acc[e] += __shfl_xor(acc[e], 1, 32);
  }
  if (lane == 0) {
    float m = acc[0]; int am = 0;
#pragma unroll
    for (int e = 1; e < E_; ++e) if (acc[e] > m) { m = acc[e]; am = e; }
    float sum = 0.f;
#pragma unroll
    for (int e = 0; e < E_; ++e) sum += __expf(acc[e] - m);
#pragma unroll
    for (int e = 0; e < E_; ++e) logits_out[(size_t)token * E_ + e] = acc[e];
    expert_id[token] = am;
    maxprob[token] = 1.f / sum;   // exp(max-max)/sum
  }
}

// ---------------- Ordered capacity scan per (b,e) via wave ballot ----------
__global__ __launch_bounds__(32) void k_scan(const int* __restrict__ expert_id,
                                             int* __restrict__ slot_token,
                                             int* __restrict__ counts,
                                             int* __restrict__ expert_index_out) {
  int b = blockIdx.x / E_, e = blockIdx.x % E_, lane = threadIdx.x;
  int running = 0;
  for (int s0 = 0; s0 < S_; s0 += 32) {
    int s = s0 + lane;
    int match = (expert_id[b * S_ + s] == e) ? 1 : 0;
    unsigned mask = (unsigned)__ballot(match);
    int prefix = __popc(mask & ((1u << lane) - 1u));
    if (match) {
      int p = running + prefix;
      expert_index_out[b * S_ + s] = (p < C_) ? e : 0;
      if (p < C_) slot_token[(b * E_ + e) * C_ + p] = s;
    }
    running += __popc(mask);
  }
  if (lane == 0) counts[b * E_ + e] = running < C_ ? running : C_;
}

// ---------------- Zero the main output region ------------------------------
__global__ __launch_bounds__(256) void k_zero(float4* __restrict__ p, int n4) {
  int i = blockIdx.x * 256 + threadIdx.x;
  if (i < n4) p[i] = make_float4(0.f, 0.f, 0.f, 0.f);
}

// ---------------- Gather tokens into compact bf16 A buffer -----------------
__global__ __launch_bounds__(256) void k_gather(const float* __restrict__ hs,
                                                const int* __restrict__ slot_token,
                                                const int* __restrict__ counts,
                                                unsigned short* __restrict__ Abf) {
  int wave = threadIdx.x >> 5, lane = threadIdx.x & 31;
  int r = blockIdx.x * 8 + wave;
  int pair = r / C_, c = r % C_;
  unsigned short* dst = Abf + (size_t)r * D_;
  if (c < counts[pair]) {
    int bb = pair / E_;
    int s = slot_token[r];
    const float* src = hs + ((size_t)bb * S_ + s) * D_;
    for (int j = 0; j < D_ / 32; ++j) dst[j * 32 + lane] = f2bf(src[j * 32 + lane]);
  } else {
    for (int j = 0; j < D_ / 32; ++j) dst[j * 32 + lane] = 0;
  }
}

// ---------------- Weight transpose + f32->bf16 (N-major for B tiles) -------
__global__ __launch_bounds__(256) void k_wtrans(const float* __restrict__ wi,
                                                const float* __restrict__ wo,
                                                unsigned short* __restrict__ WiT,
                                                unsigned short* __restrict__ WoT) {
  __shared__ float t[32][33];
  int z = blockIdx.z;
  int eidx = z & (E_ - 1);
  bool isWo = z >= E_;
  const float* src; unsigned short* dst; int R, Cc, rt, ct;
  if (!isWo) { src = wi + (size_t)eidx * D_ * F_; dst = WiT + (size_t)eidx * F_ * D_;
               R = D_; Cc = F_; rt = blockIdx.y; ct = blockIdx.x; }
  else       { src = wo + (size_t)eidx * F_ * D_; dst = WoT + (size_t)eidx * D_ * F_;
               R = F_; Cc = D_; rt = blockIdx.x; ct = blockIdx.y; }
  int tx = threadIdx.x & 31, ty = threadIdx.x >> 5;   // ty 0..7
  int r0 = rt * 32, c0 = ct * 32;
#pragma unroll
  for (int i = 0; i < 4; ++i)
    t[ty + 8 * i][tx] = src[(size_t)(r0 + ty + 8 * i) * Cc + c0 + tx];
  __syncthreads();
#pragma unroll
  for (int i = 0; i < 4; ++i)
    dst[(size_t)(c0 + ty + 8 * i) * R + r0 + tx] = f2bf(t[tx][ty + 8 * i]);
}

// ---------------- WMMA GEMM: C[512 x N] = A[512 x K] * Bt[N x K]^T ---------
// Double-buffered LDS, register prefetch, global_prefetch two steps ahead.
// mode 0: Hout = relu(A*W) in bf16   |  mode 1: out[token] = (A*W)*maxprob
__global__ __launch_bounds__(256) void k_gemm(const unsigned short* __restrict__ A,
                                              const unsigned short* __restrict__ Bt,
                                              int K, int N, int mode,
                                              unsigned short* __restrict__ Hout,
                                              float* __restrict__ out,
                                              const int* __restrict__ counts,
                                              const int* __restrict__ slot_token,
                                              const float* __restrict__ maxprob) {
  __shared__ unsigned short Al[2][128 * 32];
  __shared__ unsigned short Bl[2][128 * 32];
  int pair = blockIdx.z;
  int m0 = blockIdx.y * 128;
  int n0 = blockIdx.x * 128;
  const unsigned short* Ab = A + (size_t)pair * C_ * K;
  const unsigned short* Bb = Bt + (size_t)(pair % E_) * (size_t)N * K;
  int tid = threadIdx.x, lane = tid & 31, wave = tid >> 5;
  int wm = (wave >> 2) * 64, wn = (wave & 3) * 32;
  int lr = lane & 15, kg = lane >> 4;

  // per-thread tile-copy coordinates: 16 halfs (32B) each
  int row = tid >> 1, part = tid & 1;
  const unsigned short* gaBase = Ab + (size_t)(m0 + row) * K + part * 16;
  const unsigned short* gbBase = Bb + (size_t)(n0 + row) * K + part * 16;
  int lofs = row * 32 + part * 16;

  v8f acc[4][2];
#pragma unroll
  for (int i = 0; i < 4; ++i)
#pragma unroll
    for (int j = 0; j < 2; ++j)
      acc[i][j] = v8f{0.f, 0.f, 0.f, 0.f, 0.f, 0.f, 0.f, 0.f};

  // prologue: stage k-step 0 into buffer 0
  uint4 ra0, ra1, rb0, rb1;
  {
    const uint4* ga = (const uint4*)gaBase;
    ra0 = ga[0]; ra1 = ga[1];
    const uint4* gb = (const uint4*)gbBase;
    rb0 = gb[0]; rb1 = gb[1];
    uint4* la = (uint4*)&Al[0][lofs]; la[0] = ra0; la[1] = ra1;
    uint4* lb = (uint4*)&Bl[0][lofs]; lb[0] = rb0; lb[1] = rb1;
  }
  __syncthreads();

  int steps = K / 32;
  for (int i = 0; i < steps; ++i) {
    int buf = i & 1;
    // prefetch next tile into registers while WMMAs issue
    if (i + 1 < steps) {
      const uint4* ga = (const uint4*)(gaBase + (size_t)(i + 1) * 32);
      ra0 = ga[0]; ra1 = ga[1];
      const uint4* gb = (const uint4*)(gbBase + (size_t)(i + 1) * 32);
      rb0 = gb[0]; rb1 = gb[1];
      if (i + 2 < steps) {
        __builtin_prefetch(gaBase + (size_t)(i + 2) * 32, 0, 3);
        __builtin_prefetch(gbBase + (size_t)(i + 2) * 32, 0, 3);
      }
    }

    v16bf afr[4], bfr[2];
#pragma unroll
    for (int mt = 0; mt < 4; ++mt) {
      int r2 = wm + mt * 16 + lr;
      v8us lo = *(const v8us*)&Al[buf][r2 * 32 + kg * 8];        // K = kg*8 .. +7
      v8us hi = *(const v8us*)&Al[buf][r2 * 32 + 16 + kg * 8];   // K = 16+kg*8 .. +7
      union { v16us u; v16bf b; } cv;
      cv.u = __builtin_shufflevector(lo, hi, 0,1,2,3,4,5,6,7,8,9,10,11,12,13,14,15);
      afr[mt] = cv.b;
    }
#pragma unroll
    for (int nt = 0; nt < 2; ++nt) {
      int r2 = wn + nt * 16 + lr;                                 // r2 = N index
      v8us lo = *(const v8us*)&Bl[buf][r2 * 32 + kg * 8];        // K = kg*8 .. +7
      v8us hi = *(const v8us*)&Bl[buf][r2 * 32 + 16 + kg * 8];   // K = 16+kg*8 .. +7
      union { v16us u; v16bf b; } cv;
      cv.u = __builtin_shufflevector(lo, hi, 0,1,2,3,4,5,6,7,8,9,10,11,12,13,14,15);
      bfr[nt] = cv.b;
    }
#pragma unroll
    for (int mt = 0; mt < 4; ++mt)
#pragma unroll
      for (int nt = 0; nt < 2; ++nt)
        acc[mt][nt] = __builtin_amdgcn_wmma_f32_16x16x32_bf16(
            false, afr[mt], false, bfr[nt], (short)0, acc[mt][nt], false, false);

    // commit next tile into the other buffer; one barrier per step
    if (i + 1 < steps) {
      uint4* la = (uint4*)&Al[buf ^ 1][lofs]; la[0] = ra0; la[1] = ra1;
      uint4* lb = (uint4*)&Bl[buf ^ 1][lofs]; lb[0] = rb0; lb[1] = rb1;
      __syncthreads();
    }
  }

  int lh = lane >> 4;
  if (mode == 0) {
#pragma unroll
    for (int mt = 0; mt < 4; ++mt)
#pragma unroll
      for (int nt = 0; nt < 2; ++nt)
#pragma unroll
        for (int v = 0; v < 8; ++v) {
          int m = m0 + wm + mt * 16 + v + 8 * lh;
          int n = n0 + wn + nt * 16 + lr;
          float x = acc[mt][nt][v];
          x = x > 0.f ? x : 0.f;
          Hout[((size_t)pair * C_ + m) * (size_t)N + n] = f2bf(x);
        }
  } else {
    int cnt = counts[pair];
    int bb = pair / E_;
#pragma unroll
    for (int mt = 0; mt < 4; ++mt)
#pragma unroll
      for (int v = 0; v < 8; ++v) {
        int c = m0 + wm + mt * 16 + v + 8 * lh;
        if (c < cnt) {
          int s = slot_token[pair * C_ + c];
          float sc = maxprob[bb * S_ + s];
          float* orow = out + ((size_t)bb * S_ + s) * D_;
#pragma unroll
          for (int nt = 0; nt < 2; ++nt)
            orow[n0 + wn + nt * 16 + lr] = acc[mt][nt][v] * sc;
        }
      }
  }
}

// --------------------------------------------------------------------------
extern "C" void kernel_launch(void* const* d_in, const int* in_sizes, int n_in,
                              void* d_out, int out_size, void* d_ws, size_t ws_size,
                              hipStream_t stream) {
  const float* hs = (const float*)d_in[0];   // (B,S,D)
  const float* rw = (const float*)d_in[1];   // (D,E)
  const float* wi = (const float*)d_in[2];   // (E,D,F)
  const float* wo = (const float*)d_in[3];   // (E,F,D)

  float* out    = (float*)d_out;                 // (B,S,D)
  float* logits = out + (size_t)T_ * D_;         // (B,S,E)
  int*   eidx   = (int*)(logits + (size_t)T_ * E_);  // (B,S) int32 bits

  char* ws = (char*)d_ws;
  size_t off = 0;
  int*   expert_id  = (int*)(ws + off);   off += (size_t)T_ * 4;          // 64KB
  float* maxprob    = (float*)(ws + off); off += (size_t)T_ * 4;          // 64KB
  int*   counts     = (int*)(ws + off);   off += 4096;                    // pad
  int*   slot_token = (int*)(ws + off);   off += (size_t)ROWS * 4;        // 128KB
  off = (off + 255) & ~(size_t)255;
  unsigned short* Abf = (unsigned short*)(ws + off); off += (size_t)ROWS * D_ * 2;    // 64MB
  unsigned short* WiT = (unsigned short*)(ws + off); off += (size_t)E_ * F_ * D_ * 2; // 64MB
  unsigned short* WoT = (unsigned short*)(ws + off); off += (size_t)E_ * D_ * F_ * 2; // 64MB
  unsigned short* Hbf = (unsigned short*)(ws + off); off += (size_t)ROWS * F_ * 2;    // 256MB
  (void)ws_size; (void)in_sizes; (void)n_in; (void)out_size;

  // 1) router
  k_router<<<T_ / 8, 256, 0, stream>>>(hs, rw, logits, expert_id, maxprob);
  // 2) ordered capacity scan
  k_scan<<<PAIRS, 32, 0, stream>>>(expert_id, slot_token, counts, eidx);
  // 3) zero main output (dropped tokens stay 0)
  {
    int n4 = T_ * D_ / 4;
    k_zero<<<(n4 + 255) / 256, 256, 0, stream>>>((float4*)out, n4);
  }
  // 4) gather tokens -> bf16 A buffer
  k_gather<<<ROWS / 8, 256, 0, stream>>>(hs, slot_token, counts, Abf);
  // 5) transpose+convert weights (N-major bf16)
  k_wtrans<<<dim3(128, 32, 16), 256, 0, stream>>>(wi, wo, WiT, WoT);
  // 6) H = relu(A @ wi)   [K=D, N=F]
  k_gemm<<<dim3(F_ / 128, C_ / 128, PAIRS), 256, 0, stream>>>(
      Abf, WiT, D_, F_, 0, Hbf, nullptr, counts, slot_token, maxprob);
  // 7) out = (H @ wo) * maxprob, scattered to tokens   [K=F, N=D]
  k_gemm<<<dim3(D_ / 128, C_ / 128, PAIRS), 256, 0, stream>>>(
      Hbf, WoT, F_, D_, 1, nullptr, out, counts, slot_token, maxprob);
}
